// StructureExtractor_28278064677354
// MI455X (gfx1250) — compile-verified, matrix operand order checked by reading, not run on hardware
//
#include <hip/hip_runtime.h>
#include <hip/hip_bf16.h>

#define B_   16
#define S_   512
#define N_   64
#define H_   768
#define E_   9
#define R_   10
#define P_   2016    // N*(N-1)/2
#define H2_  384     // H/2
#define H2R_ 1536    // 2*H

typedef __attribute__((ext_vector_type(16))) __bf16 v16bf;
typedef __attribute__((ext_vector_type(8)))  float  v8f;

__device__ __forceinline__ unsigned short f2bf(float f) {
    unsigned u = __float_as_uint(f);
    unsigned r = u + 0x7FFFu + ((u >> 16) & 1u);   // round-to-nearest-even
    return (unsigned short)(r >> 16);
}

__device__ __forceinline__ unsigned pack2bf(float lo, float hi) {
    return (unsigned)f2bf(lo) | ((unsigned)f2bf(hi) << 16);
}

__device__ __forceinline__ v8f wmma_bf16(v16bf a, v16bf b, v8f c) {
    // D(f32 16x16) = A(bf16 16x32) x B(bf16 32x16) + C
    return __builtin_amdgcn_wmma_f32_16x16x32_bf16(
        false, a, false, b, (short)0, c, false, false);
}

// K-offset within a 32-wide K step for 16-bit A/B fragment element j (0..15),
// per ISA 7.12.2 16-bit 16x32 layout. Note: j<8 -> half*8 + j (contiguous),
// j>=8 -> 16 + half*8 + (j-8) (contiguous) — exploited for vector staging.
__device__ __forceinline__ int klocal(int half, int j) {
    int v = j >> 1, h = j & 1;
    return ((v >> 2) << 4) + (half << 3) + ((v & 3) << 1) + h;
}

// Convert 8 consecutive f32 (two float4s) into 8 packed bf16 (one uint4).
__device__ __forceinline__ uint4 cvt8bf(const float* __restrict__ src) {
    float4 a = ((const float4*)src)[0];
    float4 b = ((const float4*)src)[1];
    uint4 r;
    r.x = pack2bf(a.x, a.y);
    r.y = pack2bf(a.z, a.w);
    r.z = pack2bf(b.x, b.y);
    r.w = pack2bf(b.z, b.w);
    return r;
}

// ---------------------------------------------------------------------------
// Pack a row-major (K x N) f32 weight matrix into fragment-major bf16 order:
// dst[(nt*ksteps + ks)*512 + lane*16 + j]  (one v16bf per lane per fragment).
// Lane l of a B-fragment holds column n = nt*16 + (l&15); K per klocal().
// Columns >= N are zero-padded (for the 9/10-wide heads padded to 16).
// Runs once per launch; scalar form is fine.
// ---------------------------------------------------------------------------
__global__ void pack_wmma_b_kernel(const float* __restrict__ src,
                                   unsigned short* __restrict__ dst,
                                   int K, int N, int ntiles, int ksteps) {
    int t = blockIdx.x * blockDim.x + threadIdx.x;
    int total = ntiles * ksteps * 32;
    if (t >= total) return;
    int lane = t & 31;
    int frag = t >> 5;
    int ks = frag % ksteps;
    int nt = frag / ksteps;
    int nl = lane & 15, half = lane >> 4;
    int n = nt * 16 + nl;
    unsigned short* o = dst + (size_t)frag * 512 + lane * 16;
    #pragma unroll
    for (int j = 0; j < 16; ++j) {
        int k = ks * 32 + klocal(half, j);
        float v = (n < N && k < K) ? src[(size_t)k * N + n] : 0.0f;
        o[j] = f2bf(v);
    }
}

// triu_indices(N, k=1) row-major order -> (i, j) per pair index.
__global__ void build_pairs_kernel(int* __restrict__ pairs) {
    int p = blockIdx.x * blockDim.x + threadIdx.x;
    if (p >= P_) return;
    int i = 0, rem = p, avail = N_ - 1;
    while (rem >= avail) { rem -= avail; --avail; ++i; }
    int j = i + 1 + rem;
    pairs[2 * p] = i;
    pairs[2 * p + 1] = j;
}

// ---------------------------------------------------------------------------
// Span mean pooling: entity_repr[b,n,:] = mean(seq[b, start:end, :]).
// ---------------------------------------------------------------------------
__global__ void span_pool_kernel(const float* __restrict__ seq,
                                 const int* __restrict__ spans,
                                 float* __restrict__ entRepr) {
    int bi = blockIdx.x;            // b*N + n
    int bb = bi >> 6;
    int start = spans[bi * 2];
    int end   = spans[bi * 2 + 1];
    if (start < 0) start = 0;
    if (end > S_) end = S_;
    int cnt = end - start; if (cnt < 1) cnt = 1;
    int tid = threadIdx.x;
    float a0 = 0.f, a1 = 0.f, a2 = 0.f;
    const float* base = seq + (size_t)bb * S_ * H_;
    for (int s = start; s < end; ++s) {
        const float* row = base + (size_t)s * H_;
        a0 += row[tid];
        a1 += row[tid + 256];
        a2 += row[tid + 512];
    }
    float inv = 1.0f / (float)cnt;
    float* o = entRepr + (size_t)bi * H_;
    o[tid]       = a0 * inv;
    o[tid + 256] = a1 * inv;
    o[tid + 512] = a2 * inv;
}

// ---------------------------------------------------------------------------
// Entity MLP: logits = relu(seq @ W1e + b1e) @ W2e + b2e.
// One block per 16-row M-tile (512 blocks, 8 waves). K outer / N-tile inner
// so a single A fragment is live per K step (no spills).
// ---------------------------------------------------------------------------
__global__ void entity_mlp_kernel(const float* __restrict__ seq,
                                  const unsigned short* __restrict__ w1p,
                                  const float* __restrict__ b1e,
                                  const unsigned short* __restrict__ w2p,
                                  const float* __restrict__ b2e,
                                  float* __restrict__ logits) {
    __shared__ __align__(32) unsigned short Afrag[24 * 32 * 16]; // 16x768 bf16, frag-major
    __shared__ __align__(32) unsigned short Hid[16 * H2_];       // 16x384 bf16, row-major
    const int tid  = threadIdx.x;
    const int row0 = blockIdx.x * 16;

    // Stage A-tile: vectorized (two 8-wide runs per lane-fragment).
    for (int i = 0; i < 3; ++i) {
        int idx = tid + i * 256;               // 768 lane-fragments
        int ks = idx >> 5, ln = idx & 31;
        int m = ln & 15, half = ln >> 4;
        const float* sr = seq + (size_t)(row0 + m) * H_ + ks * 32 + half * 8;
        uint4* o = (uint4*)&Afrag[idx * 16];
        o[0] = cvt8bf(sr);        // k = ks*32 + half*8 + [0,8)
        o[1] = cvt8bf(sr + 16);   // k = ks*32 + 16 + half*8 + [0,8)
    }
    __syncthreads();

    const int wave = tid >> 5, lane = tid & 31;
    const int nl = lane & 15, half = lane >> 4;
    const v16bf* Ap = reinterpret_cast<const v16bf*>(Afrag);
    const v16bf* Bp = reinterpret_cast<const v16bf*>(w1p);

    // GEMM1: each wave owns 3 of 24 N-tiles; K = 768 -> 24 steps.
    v8f acc[3];
    #pragma unroll
    for (int t = 0; t < 3; ++t)
        acc[t] = (v8f){0.f,0.f,0.f,0.f,0.f,0.f,0.f,0.f};

    #pragma unroll 1
    for (int ks = 0; ks < 24; ++ks) {
        v16bf a = Ap[ks * 32 + lane];
        #pragma unroll
        for (int t = 0; t < 3; ++t) {
            int nt = wave + t * 8;
            v16bf b = Bp[(size_t)(nt * 24 + ks) * 32 + lane];
            acc[t] = wmma_bf16(a, b, acc[t]);
        }
    }

    #pragma unroll
    for (int t = 0; t < 3; ++t) {
        int nt = wave + t * 8;
        float bias = b1e[nt * 16 + nl];
        #pragma unroll
        for (int r = 0; r < 8; ++r) {
            int m = r + 8 * half;              // C layout: VGPR r -> M = r + 8*half
            float v = acc[t][r] + bias;
            v = v > 0.f ? v : 0.f;
            Hid[m * H2_ + nt * 16 + nl] = f2bf(v);
        }
    }
    __syncthreads();

    // GEMM2 head (K=384, N=9 padded to 16), wave 0 only (wave-uniform EXEC).
    if (wave == 0) {
        const v16bf* B2 = reinterpret_cast<const v16bf*>(w2p);
        v8f acc2 = {0.f,0.f,0.f,0.f,0.f,0.f,0.f,0.f};
        #pragma unroll 1
        for (int ks = 0; ks < 12; ++ks) {
            int base = nl * H2_ + ks * 32 + half * 8;
            union { v16bf v; uint4 q[2]; } au;
            au.q[0] = *(const uint4*)&Hid[base];
            au.q[1] = *(const uint4*)&Hid[base + 16];
            v16bf b = B2[(size_t)ks * 32 + lane];
            acc2 = wmma_bf16(au.v, b, acc2);
        }
        if (nl < E_) {
            float bias = b2e[nl];
            #pragma unroll
            for (int r = 0; r < 8; ++r) {
                int m = r + 8 * half;
                logits[(size_t)(row0 + m) * E_ + nl] = acc2[r] + bias;
            }
        }
    }
}

// ---------------------------------------------------------------------------
// Relation MLP: relu(pair_repr @ W1r + b1r) @ W2r + b2r, fused.
// pair_repr never materialized: 16 pair rows (concat of two entity_repr rows)
// are gathered into LDS in fragment order. K outer / N-tile inner keeps one
// A fragment live; 6 v8f accumulators per wave. A-fragment LDS buffer is
// reused as the 16x768 hidden buffer after a barrier (stays under 64 KB).
// ---------------------------------------------------------------------------
__global__ void relation_mlp_kernel(const float* __restrict__ entRepr,
                                    const int* __restrict__ pairs,
                                    const unsigned short* __restrict__ w1p,
                                    const float* __restrict__ b1r,
                                    const unsigned short* __restrict__ w2p,
                                    const float* __restrict__ b2r,
                                    float* __restrict__ relLogits) {
    __shared__ __align__(32) unsigned short Buf[48 * 32 * 16]; // 49152 B
    const int tid   = threadIdx.x;
    const int bb    = blockIdx.x / 126;          // 126 = 2016/16 tiles per batch
    const int pbase = (blockIdx.x % 126) * 16;

    // Stage A-tile: 16 pair rows x K=1536, fragment-major bf16, vectorized.
    // A 32-wide K block never straddles the two concatenated entity rows.
    for (int i = 0; i < 6; ++i) {
        int idx = tid + i * 256;                 // 1536 lane-fragments
        int ks = idx >> 5, ln = idx & 31;
        int m = ln & 15, half = ln >> 4;
        int p = pbase + m;
        int ei = pairs[2 * p], ej = pairs[2 * p + 1];
        int ent = (ks < 24) ? ei : ej;
        int ko  = (ks < 24) ? ks * 32 : (ks - 24) * 32;
        const float* sr = entRepr + (size_t)(bb * N_ + ent) * H_ + ko + half * 8;
        uint4* o = (uint4*)&Buf[idx * 16];
        o[0] = cvt8bf(sr);
        o[1] = cvt8bf(sr + 16);
    }
    __syncthreads();

    const int wave = tid >> 5, lane = tid & 31;
    const int nl = lane & 15, half = lane >> 4;
    const v16bf* Ap = reinterpret_cast<const v16bf*>(Buf);
    const v16bf* Bp = reinterpret_cast<const v16bf*>(w1p);

    v8f acc[6];
    #pragma unroll
    for (int t = 0; t < 6; ++t)
        acc[t] = (v8f){0.f,0.f,0.f,0.f,0.f,0.f,0.f,0.f};

    // GEMM3: 48 N-tiles across 8 waves; K = 1536 -> 48 steps.
    #pragma unroll 1
    for (int ks = 0; ks < 48; ++ks) {
        v16bf a = Ap[ks * 32 + lane];
        #pragma unroll
        for (int t = 0; t < 6; ++t) {
            int nt = wave + t * 8;
            v16bf b = Bp[(size_t)(nt * 48 + ks) * 32 + lane];
            acc[t] = wmma_bf16(a, b, acc[t]);
        }
    }
    __syncthreads();   // everyone done reading A-fragments; Buf becomes Hid

    #pragma unroll
    for (int t = 0; t < 6; ++t) {
        int nt = wave + t * 8;
        float bias = b1r[nt * 16 + nl];
        #pragma unroll
        for (int r = 0; r < 8; ++r) {
            int m = r + 8 * half;
            float v = acc[t][r] + bias;
            v = v > 0.f ? v : 0.f;
            Buf[m * H_ + nt * 16 + nl] = f2bf(v);   // Hid: 16 x 768 row-major
        }
    }
    __syncthreads();

    // GEMM4 head (K=768, N=10 padded to 16), wave 0 only.
    if (wave == 0) {
        const v16bf* B2 = reinterpret_cast<const v16bf*>(w2p);
        v8f acc2 = {0.f,0.f,0.f,0.f,0.f,0.f,0.f,0.f};
        #pragma unroll 1
        for (int ks = 0; ks < 24; ++ks) {
            int base = nl * H_ + ks * 32 + half * 8;
            union { v16bf v; uint4 q[2]; } au;
            au.q[0] = *(const uint4*)&Buf[base];
            au.q[1] = *(const uint4*)&Buf[base + 16];
            v16bf b = B2[(size_t)ks * 32 + lane];
            acc2 = wmma_bf16(au.v, b, acc2);
        }
        if (nl < R_) {
            float bias = b2r[nl];
            #pragma unroll
            for (int r = 0; r < 8; ++r) {
                int m = r + 8 * half;
                int g = bb * P_ + pbase + m;
                relLogits[(size_t)g * R_ + nl] = acc2[r] + bias;
            }
        }
    }
}

// ---------------------------------------------------------------------------
extern "C" void kernel_launch(void* const* d_in, const int* in_sizes, int n_in,
                              void* d_out, int out_size, void* d_ws, size_t ws_size,
                              hipStream_t stream) {
    const float* seq   = (const float*)d_in[0];
    // d_in[1]: attention_mask (unused by reference math)
    const int*   spans = (const int*)d_in[2];
    const float* W1e   = (const float*)d_in[3];
    const float* b1e   = (const float*)d_in[4];
    const float* W2e   = (const float*)d_in[5];
    const float* b2e   = (const float*)d_in[6];
    const float* W1r   = (const float*)d_in[7];
    const float* b1r   = (const float*)d_in[8];
    const float* W2r   = (const float*)d_in[9];
    const float* b2r   = (const float*)d_in[10];

    float* out       = (float*)d_out;
    float* entLogits = out;                               // B*S*E   = 73728
    float* entRepr   = out + (size_t)B_ * S_ * E_;        // B*N*H   = 786432
    float* relLogits = entRepr + (size_t)B_ * N_ * H_;    // B*P*R   = 322560

    // Workspace: fragment-packed bf16 weights + pair table (~3 MB).
    unsigned short* w1e_p = (unsigned short*)d_ws;        // 24*24 frags
    unsigned short* w2e_p = w1e_p + (size_t)576  * 512;   //  1*12 frags
    unsigned short* w1r_p = w2e_p + (size_t)12   * 512;   // 48*48 frags
    unsigned short* w2r_p = w1r_p + (size_t)2304 * 512;   //  1*24 frags
    int* pairs            = (int*)(w2r_p + (size_t)24 * 512);

    pack_wmma_b_kernel<<<72,  256, 0, stream>>>(W1e, w1e_p, H_,   H2_, 24, 24);
    pack_wmma_b_kernel<<<2,   256, 0, stream>>>(W2e, w2e_p, H2_,  E_,   1, 12);
    pack_wmma_b_kernel<<<288, 256, 0, stream>>>(W1r, w1r_p, H2R_, H_,  48, 48);
    pack_wmma_b_kernel<<<3,   256, 0, stream>>>(W2r, w2r_p, H_,   R_,   1, 24);
    build_pairs_kernel<<<(P_ + 255) / 256, 256, 0, stream>>>(pairs);

    span_pool_kernel<<<B_ * N_, 256, 0, stream>>>(seq, spans, entRepr);
    entity_mlp_kernel<<<(B_ * S_) / 16, 256, 0, stream>>>(
        seq, w1e_p, b1e, w2e_p, b2e, entLogits);
    relation_mlp_kernel<<<(B_ * P_) / 16, 256, 0, stream>>>(
        entRepr, pairs, w1r_p, b1r, w2r_p, b2r, relLogits);
}
